// Decoder_20564303413277
// MI455X (gfx1250) — matrix operand order repaired and checked by reading, not compiled, not run on hardware
//
#include <hip/hip_runtime.h>
#include <hip/hip_bf16.h>

// Decoder (2-layer LSTM + Luong attention + vocab projection) for MI455X.
//   - f16 WMMA (v_wmma_f32_16x16x32_f16) everywhere, fp32 accumulate/state.
//   - Weights packed once per launch into WMMA-B-fragment layout in d_ws.
//   - One persistent 1024-thread workgroup runs the 64-step recurrence.
//     Gate GEMMs are split into two K-passes (acc[4] each) to stay under the
//     128-VGPR/wave budget of a 32-wave workgroup (no scratch spills).
//   - Vocab projection (2048 x 32000, K=512, 67 GFLOP) batched outside the
//     time loop; 64-row x 128-col tiles, B double-buffered, B reused x4.
// Workspace requirement: ~46.8 MB.

#define VSZ 32000
#define ESZ 512
#define HSZ 512
#define BSZ 32
#define TSZ 64
#define SSZ 64

typedef __attribute__((ext_vector_type(16))) _Float16 v16h;
typedef __attribute__((ext_vector_type(8)))  float    v8f;

// ---- WMMA fragment index maps (16-bit layouts; A/B k order is this file's own) ----
// A 16x32:  lane l -> (m = l&15, hi = l>>4); element j -> K = hi*16 + j (this file's order, not the ISA table's)
// B 32x16:  lane l -> (n = l&15, hi = l>>4); element j -> K = hi*16 + j (this file's order, not the ISA table's)
// C 16x16:  lane l -> (n = l&15, hi = l>>4); element j -> M = hi*8 + j

__device__ __forceinline__ void ka_inv(int K, int& hi, int& j) {
  hi = K >> 4; j = K & 15;
}

__device__ __forceinline__ v8f wmma16(v16h a, v16h b, v8f c) {
  return __builtin_amdgcn_wmma_f32_16x16x32_f16(false, a, false, b,
                                                (short)0, c, false, false);
}

__device__ __forceinline__ v8f vzero8() {
  v8f z = {0.f, 0.f, 0.f, 0.f, 0.f, 0.f, 0.f, 0.f};
  return z;
}

__device__ __forceinline__ float sigf(float x) { return 1.0f / (1.0f + __expf(-x)); }

// Store X[row][k] (fp32) into LDS packed-A-fragment layout:
// xp[ ((mi*48 + kc)*32 + lane)*16 + j ], lane = hi*16 + (row&15)
__device__ __forceinline__ void xpack_store(_Float16* xp, int row, int k, float val) {
  int mi = row >> 4, ml = row & 15;
  int kc = k >> 5, rem = k & 31, hi, j;
  ka_inv(rem, hi, j);
  int lane = (hi << 4) | ml;
  xp[((((mi * 48) + kc) << 5) + lane) * 16 + j] = (_Float16)val;
}

// ---------------------------------------------------------------------------
// Pack W (row-major [N][k1], optionally concat W2 [N][k2] along K) into
// B-fragment layout: dst[((kc*2+hi)*N + n)*16 + j] = W[n][kc*32 + hi*16 + j]
// ---------------------------------------------------------------------------
__global__ void pack_b_kernel(const float* __restrict__ W1, int k1,
                              const float* __restrict__ W2, int k2,
                              int N, _Float16* __restrict__ dst) {
  long total = (long)(k1 + k2) * (long)N;
  for (long idx = (long)blockIdx.x * blockDim.x + threadIdx.x; idx < total;
       idx += (long)gridDim.x * blockDim.x) {
    int  j  = (int)(idx & 15);
    long t2 = idx >> 4;
    int  n  = (int)(t2 % N);
    long t3 = t2 / N;
    int  hi = (int)(t3 & 1);
    int  kc = (int)(t3 >> 1);
    int  k  = kc * 32 + hi * 16 + j;
    float v = (k < k1) ? W1[(long)n * k1 + k] : W2[(long)n * k2 + (k - k1)];
    dst[idx] = (_Float16)v;
  }
}

__global__ void bias_kernel(const float* __restrict__ bi0, const float* __restrict__ bh0,
                            const float* __restrict__ bi1, const float* __restrict__ bh1,
                            float* __restrict__ dst) {
  int i = blockIdx.x * blockDim.x + threadIdx.x;
  if (i < 2048)      dst[i] = bi0[i] + bh0[i];
  else if (i < 4096) dst[i] = bi1[i - 2048] + bh1[i - 2048];
}

// ---------------------------------------------------------------------------
// Persistent recurrent kernel: 1 block, 1024 threads (32 waves).
// Wave w owns row-tile mi = w&1 and hidden-slice hj = w>>1 (32 cols); the
// i/f/g/o gate tiles of one 16-col slice are computed per K-pass so the
// LSTM cell combine happens entirely in registers.
// ---------------------------------------------------------------------------
__global__ __launch_bounds__(1024) void decoder_recurrent(
    const float* __restrict__ src, const int* __restrict__ trg,
    const float* __restrict__ embW, const float* __restrict__ bias2,
    const _Float16* __restrict__ Bg0, const _Float16* __restrict__ Bg1,
    const _Float16* __restrict__ Batt, const float* __restrict__ attb,
    const float* __restrict__ ifeed,
    float* __restrict__ hstate, float* __restrict__ cstate,
    float* __restrict__ hhat, _Float16* __restrict__ Apack,
    float* __restrict__ out_tail) {
  extern __shared__ unsigned char smem[];
  _Float16* xp   = (_Float16*)smem;                   // 2*48*32*16 f16 = 98304 B
  float*    hcur = (float*)(smem + 98304);            // 32 * 520 f32  = 66560 B
  float*    smrw = (float*)(smem + 98304 + 66560);    // 32 * 64  f32  =  8192 B

  const int tid  = threadIdx.x;
  const int lane = tid & 31;
  const int wave = tid >> 5;
  const int nl   = lane & 15;
  const int hi   = lane >> 4;
  const int mi   = wave & 1;
  const int hj   = wave >> 1;   // 0..15

  // Per-lane GEMM base pointers (per-kc strides are induction adds,
  // per-fragment deltas are load-immediate offsets).
  const _Float16* ab0 = xp + mi * 24576 + lane * 16;                     // +512/kc
  const _Float16* bg0 = Bg0 + ((long)hi * 2048 + (hj << 5) + nl) * 16;   // +65536/kc
  const _Float16* bg1 = Bg1 + ((long)hi * 2048 + (hj << 5) + nl) * 16;   // +65536/kc
  const _Float16* bat = Batt + ((long)hi * 512 + (hj << 4) + nl) * 16;   // +16384/kc

  // init states: h=c=0, hhat = broadcast(input_feed_init)
  for (int i = tid; i < 2 * BSZ * HSZ; i += 1024) { hstate[i] = 0.f; cstate[i] = 0.f; }
  for (int i = tid; i < BSZ * HSZ; i += 1024)     hhat[i] = ifeed[i & (HSZ - 1)];
  __threadfence();
  __syncthreads();

  const float scale = 0.044194173824159216f;  // 1/sqrt(512)

  for (int t = 0; t < TSZ; ++t) {
    // ---- Phase 1: X0 = [x_emb | hhat | h0]  (32 x 1536) packed in LDS
    for (int idx = tid; idx < BSZ * 384; idx += 1024) {
      int b = idx / 384, k = (idx - b * 384) << 2;
      float4 v;
      if (k < 512)       v = *(const float4*)&embW[(long)trg[b * TSZ + t] * ESZ + k];
      else if (k < 1024) v = *(const float4*)&hhat[b * HSZ + (k - 512)];
      else               v = *(const float4*)&hstate[b * HSZ + (k - 1024)];
      xpack_store(xp, b, k, v.x);
      xpack_store(xp, b, k + 1, v.y);
      xpack_store(xp, b, k + 2, v.z);
      xpack_store(xp, b, k + 3, v.w);
    }
    __syncthreads();

    // ---- Phase 2+3: layer-0 gates GEMM (two K-passes) + LSTM cell 0
    float hn0[2][8];
#pragma clang loop unroll(disable)
    for (int hc = 0; hc < 2; ++hc) {
      v8f acc[4];
#pragma unroll
      for (int q = 0; q < 4; ++q) acc[q] = vzero8();
      const _Float16* ap = ab0;
      const _Float16* bp = bg0 + hc * 256;   // +16 cols
      for (int kc = 0; kc < 48; ++kc) {
        v16h af = *(const v16h*)ap;
        v16h b0 = *(const v16h*)bp;
        v16h b1 = *(const v16h*)(bp + 8192);    // gate f (+512 cols)
        v16h b2 = *(const v16h*)(bp + 16384);   // gate g
        v16h b3 = *(const v16h*)(bp + 24576);   // gate o
        acc[0] = wmma16(af, b0, acc[0]);
        acc[1] = wmma16(af, b1, acc[1]);
        acc[2] = wmma16(af, b2, acc[2]);
        acc[3] = wmma16(af, b3, acc[3]);
        ap += 512;
        bp += 65536;
      }
      int hcol = (((hj << 1) + hc) << 4) + nl;
      float bI = bias2[hcol],        bF = bias2[512 + hcol];
      float bG = bias2[1024 + hcol], bO = bias2[1536 + hcol];
#pragma unroll
      for (int j = 0; j < 8; ++j) {
        int b = (mi << 4) + (hi << 3) + j;
        float iv = sigf(acc[0][j] + bI);
        float fv = sigf(acc[1][j] + bF);
        float gv = tanhf(acc[2][j] + bG);
        float ov = sigf(acc[3][j] + bO);
        float co = cstate[b * HSZ + hcol];
        float cn = fv * co + iv * gv;
        float hn = ov * tanhf(cn);
        cstate[b * HSZ + hcol] = cn;
        hstate[b * HSZ + hcol] = hn;
        hn0[hc][j] = hn;
      }
    }
    __syncthreads();  // all xp reads of layer-0 done; now safe to overwrite
    // X1 = [h_new0 | h1_old]
#pragma unroll
    for (int hc = 0; hc < 2; ++hc) {
      int hcol = (((hj << 1) + hc) << 4) + nl;
#pragma unroll
      for (int j = 0; j < 8; ++j)
        xpack_store(xp, (mi << 4) + (hi << 3) + j, hcol, hn0[hc][j]);
    }
    for (int idx = tid; idx < BSZ * 128; idx += 1024) {
      int b = idx >> 7, h = (idx & 127) << 2;
      float4 v = *(const float4*)&hstate[BSZ * HSZ + b * HSZ + h];
      xpack_store(xp, b, 512 + h, v.x);
      xpack_store(xp, b, 513 + h, v.y);
      xpack_store(xp, b, 514 + h, v.z);
      xpack_store(xp, b, 515 + h, v.w);
    }
    __syncthreads();

    // ---- Phase 4: layer-1 gates GEMM (two K-passes) + LSTM cell 1
    float hn1[2][8];
#pragma clang loop unroll(disable)
    for (int hc = 0; hc < 2; ++hc) {
      v8f acc[4];
#pragma unroll
      for (int q = 0; q < 4; ++q) acc[q] = vzero8();
      const _Float16* ap = ab0;
      const _Float16* bp = bg1 + hc * 256;
      for (int kc = 0; kc < 32; ++kc) {
        v16h af = *(const v16h*)ap;
        v16h b0 = *(const v16h*)bp;
        v16h b1 = *(const v16h*)(bp + 8192);
        v16h b2 = *(const v16h*)(bp + 16384);
        v16h b3 = *(const v16h*)(bp + 24576);
        acc[0] = wmma16(af, b0, acc[0]);
        acc[1] = wmma16(af, b1, acc[1]);
        acc[2] = wmma16(af, b2, acc[2]);
        acc[3] = wmma16(af, b3, acc[3]);
        ap += 512;
        bp += 65536;
      }
      int hcol = (((hj << 1) + hc) << 4) + nl;
      float bI = bias2[2048 + hcol],        bF = bias2[2048 + 512 + hcol];
      float bG = bias2[2048 + 1024 + hcol], bO = bias2[2048 + 1536 + hcol];
#pragma unroll
      for (int j = 0; j < 8; ++j) {
        int b = (mi << 4) + (hi << 3) + j;
        float iv = sigf(acc[0][j] + bI);
        float fv = sigf(acc[1][j] + bF);
        float gv = tanhf(acc[2][j] + bG);
        float ov = sigf(acc[3][j] + bO);
        float co = cstate[BSZ * HSZ + b * HSZ + hcol];
        float cn = fv * co + iv * gv;
        float hn = ov * tanhf(cn);
        cstate[BSZ * HSZ + b * HSZ + hcol] = cn;
        hstate[BSZ * HSZ + b * HSZ + hcol] = hn;
        hn1[hc][j] = hn;
      }
    }
    __syncthreads();  // all xp reads of layer-1 done
    // X2[:,0:512] = h_t ; hcur (fp32) = h_t
#pragma unroll
    for (int hc = 0; hc < 2; ++hc) {
      int hcol = (((hj << 1) + hc) << 4) + nl;
#pragma unroll
      for (int j = 0; j < 8; ++j) {
        int b = (mi << 4) + (hi << 3) + j;
        xpack_store(xp, b, hcol, hn1[hc][j]);
        hcur[b * 520 + hcol] = hn1[hc][j];
      }
    }
    __syncthreads();

    // ---- Phase 5: attention (fp32 VALU; tiny)
    for (int p = tid; p < BSZ * SSZ; p += 1024) {
      int b = p >> 6, s = p & 63;
      const float4* sp = (const float4*)&src[(long)((b << 6) + s) << 9];
      const float4* hp = (const float4*)&hcur[b * 520];
      float d = 0.f;
      for (int h = 0; h < HSZ / 4; ++h) {
        float4 x = sp[h], y = hp[h];
        d += x.x * y.x + x.y * y.y + x.z * y.z + x.w * y.w;
      }
      smrw[p] = d * scale;
    }
    __syncthreads();
    {  // softmax: wave b handles batch row b (2 scores per lane)
      int b = wave;
      float v0 = smrw[(b << 6) + lane], v1 = smrw[(b << 6) + lane + 32];
      float mx = fmaxf(v0, v1);
#pragma unroll
      for (int o = 16; o; o >>= 1) mx = fmaxf(mx, __shfl_xor(mx, o, 32));
      float e0 = __expf(v0 - mx), e1 = __expf(v1 - mx);
      float sum = e0 + e1;
#pragma unroll
      for (int o = 16; o; o >>= 1) sum += __shfl_xor(sum, o, 32);
      float inv = 1.f / sum;
      smrw[(b << 6) + lane] = e0 * inv;
      smrw[(b << 6) + lane + 32] = e1 * inv;
    }
    __syncthreads();
    // context: c_att[b][h] -> X2[:,512:1024]
    for (int idx = tid; idx < BSZ * 128; idx += 1024) {
      int b = idx >> 7, h = (idx & 127) << 2;
      float4 a = {0.f, 0.f, 0.f, 0.f};
      for (int s = 0; s < SSZ; ++s) {
        float p = smrw[(b << 6) + s];
        float4 sv = *(const float4*)&src[((long)((b << 6) + s) << 9) + h];
        a.x += p * sv.x; a.y += p * sv.y; a.z += p * sv.z; a.w += p * sv.w;
      }
      xpack_store(xp, b, 512 + h, a.x);
      xpack_store(xp, b, 513 + h, a.y);
      xpack_store(xp, b, 514 + h, a.z);
      xpack_store(xp, b, 515 + h, a.w);
    }
    __syncthreads();

    // ---- Phase 6: hhat = tanh([h_t|c_att] @ attW^T + attb): [32 x 512], K=1024
    {
      v8f a0 = vzero8(), a1 = vzero8();
      const _Float16* ap = ab0;
      const _Float16* bp = bat;
      for (int kc = 0; kc < 32; ++kc) {
        v16h af = *(const v16h*)ap;
        v16h b0 = *(const v16h*)bp;
        v16h b1 = *(const v16h*)(bp + 4096);   // +256 columns
        a0 = wmma16(af, b0, a0);
        a1 = wmma16(af, b1, a1);
        ap += 512;
        bp += 16384;
      }
#pragma unroll
      for (int q = 0; q < 2; ++q) {
        int col = ((hj + q * 16) << 4) + nl;
        float bb = attb[col];
        v8f& ac = q ? a1 : a0;
#pragma unroll
        for (int j = 0; j < 8; ++j) {
          int b = (mi << 4) + (hi << 3) + j;
          float hh = tanhf(ac[j] + bb);
          hhat[b * HSZ + col] = hh;
          int r = (t << 5) + b;           // fc GEMM A-row
          int kc2 = col >> 5, rem = col & 31, h2, j2;
          ka_inv(rem, h2, j2);
          Apack[(((long)r * 16 + kc2) * 2 + h2) * 16 + j2] = (_Float16)hh;
        }
      }
    }
    __threadfence();
    __syncthreads();
  }  // t loop

  // ---- final states: out_tail = [hF | cF | hhatF]
  for (int i = tid; i < 2 * BSZ * HSZ; i += 1024) out_tail[i] = hstate[i];
  for (int i = tid; i < 2 * BSZ * HSZ; i += 1024) out_tail[2 * BSZ * HSZ + i] = cstate[i];
  for (int i = tid; i < BSZ * HSZ; i += 1024)     out_tail[4 * BSZ * HSZ + i] = hhat[i];
}

// ---------------------------------------------------------------------------
// Batched vocab projection: [2048 x 32000] = Apack[2048 x 512] @ Bfc + fc_b.
// Block = 256 threads (8 waves): 64-row x 128-col tile. Wave w owns column
// tile w and all 4 row tiles, so each B fragment feeds 4 WMMAs. The global
// B fragment is double-buffered; A comes from LDS just-in-time.
// ---------------------------------------------------------------------------
__global__ __launch_bounds__(256) void fc_gemm(
    const _Float16* __restrict__ Apack, const _Float16* __restrict__ Bfc,
    const float* __restrict__ fcb, float* __restrict__ out) {
  extern __shared__ _Float16 Atile[];  // 64 rows * 512 halfs = 64 KB
  const int tid = threadIdx.x, lane = tid & 31, wave = tid >> 5;
  const int nl = lane & 15, hi = lane >> 4;
  const int r0 = blockIdx.y << 6;   // 64 rows per block
  const int n0 = blockIdx.x << 7;   // 128 cols per block

  {  // cooperative 64 KB copy (packed rows are 512 contiguous halfs)
    const uint4* sp = (const uint4*)(Apack + (long)r0 * 512);
    uint4* dp = (uint4*)Atile;
    for (int i = tid; i < 4096; i += 256) dp[i] = sp[i];
  }
  __syncthreads();

  // A frag (row tile q, k-step kc): Atile[(q*16+nl)*512 + kc*32 + hi*16]
  const _Float16* abase = Atile + nl * 512 + hi * 16;          // +8192/row-tile
  const _Float16* bbase = Bfc + ((long)hi * VSZ + n0 + (wave << 4) + nl) * 16;  // +1024000/kc

  v8f acc[4];
#pragma unroll
  for (int q = 0; q < 4; ++q) acc[q] = vzero8();

  v16h b_cur = *(const v16h*)bbase;
  for (int kc = 0; kc < 16; ++kc) {
    v16h b_nxt = b_cur;
    if (kc < 15) b_nxt = *(const v16h*)(bbase + (long)(kc + 1) * 1024000);
    const _Float16* ak = abase + kc * 32;
    v16h a0 = *(const v16h*)ak;
    v16h a1 = *(const v16h*)(ak + 8192);
    v16h a2 = *(const v16h*)(ak + 16384);
    v16h a3 = *(const v16h*)(ak + 24576);
    acc[0] = wmma16(a0, b_cur, acc[0]);
    acc[1] = wmma16(a1, b_cur, acc[1]);
    acc[2] = wmma16(a2, b_cur, acc[2]);
    acc[3] = wmma16(a3, b_cur, acc[3]);
    b_cur = b_nxt;
  }

  const int v = n0 + (wave << 4) + nl;
  const float bias = fcb[v];
#pragma unroll
  for (int q = 0; q < 4; ++q) {
#pragma unroll
    for (int j = 0; j < 8; ++j) {
      int r = r0 + (q << 4) + (hi << 3) + j;
      int b = r & 31, t = r >> 5;
      out[((long)(b << 6) + t) * VSZ + v] = acc[q][j] + bias;
    }
  }
}

// ---------------------------------------------------------------------------
extern "C" void kernel_launch(void* const* d_in, const int* in_sizes, int n_in,
                              void* d_out, int out_size, void* d_ws, size_t ws_size,
                              hipStream_t stream) {
  const float* src   = (const float*)d_in[0];
  const int*   trg   = (const int*)d_in[1];
  const float* embW  = (const float*)d_in[2];
  const float* Wih0  = (const float*)d_in[3];
  const float* Whh0  = (const float*)d_in[4];
  const float* bih0  = (const float*)d_in[5];
  const float* bhh0  = (const float*)d_in[6];
  const float* Wih1  = (const float*)d_in[7];
  const float* Whh1  = (const float*)d_in[8];
  const float* bih1  = (const float*)d_in[9];
  const float* bhh1  = (const float*)d_in[10];
  const float* attW  = (const float*)d_in[11];
  const float* attb  = (const float*)d_in[12];
  const float* fcW   = (const float*)d_in[13];
  const float* fcb   = (const float*)d_in[14];
  const float* ifeed = (const float*)d_in[15];
  float* out = (float*)d_out;

  unsigned char* ws = (unsigned char*)d_ws;
  _Float16* Bg0   = (_Float16*)(ws + 0);          // 1536*2048*2 = 6291456
  _Float16* Bg1   = (_Float16*)(ws + 6291456);    // 1024*2048*2 = 4194304
  _Float16* Batt  = (_Float16*)(ws + 10485760);   // 1024*512*2  = 1048576
  _Float16* Bfc   = (_Float16*)(ws + 11534336);   // 512*32000*2 = 32768000
  _Float16* Apack = (_Float16*)(ws + 44302336);   // 2048*512*2  = 2097152
  float* hstate   = (float*)(ws + 46399488);      // 131072
  float* cstate   = (float*)(ws + 46530560);      // 131072
  float* hhat     = (float*)(ws + 46661632);      // 65536
  float* bias2    = (float*)(ws + 46727168);      // 16384
  // total: 46743552 bytes (~46.8 MB)

  pack_b_kernel<<<2048, 256, 0, stream>>>(Wih0, 1024, Whh0, 512, 2048, Bg0);
  pack_b_kernel<<<2048, 256, 0, stream>>>(Wih1, 512, Whh1, 512, 2048, Bg1);
  pack_b_kernel<<<1024, 256, 0, stream>>>(attW, 1024, attW, 0, 512, Batt);
  pack_b_kernel<<<4096, 256, 0, stream>>>(fcW, 512, fcW, 0, VSZ, Bfc);
  bias_kernel<<<16, 256, 0, stream>>>(bih0, bhh0, bih1, bhh1, bias2);

  decoder_recurrent<<<1, 1024, 98304 + 66560 + 8192, stream>>>(
      src, trg, embW, bias2, Bg0, Bg1, Batt, attb, ifeed,
      hstate, cstate, hhat, Apack, out + (long)BSZ * TSZ * VSZ);

  fc_gemm<<<dim3(250, 32), 256, 65536, stream>>>(Apack, Bfc, fcb, out);
}
